// Decoder_32925219291461
// MI455X (gfx1250) — compile-verified, matrix operand order checked by reading, not run on hardware
//
#include <hip/hip_runtime.h>
#include <hip/hip_bf16.h>

// Problem constants (match reference)
#define BB 64
#define TT 48
#define NN 128
#define FF 256
#define LL 2

typedef __attribute__((ext_vector_type(16))) __bf16 v16bf;
typedef __attribute__((ext_vector_type(8)))  float  v8f;

// ---------------------------------------------------------------------------
// WMMA fragment helpers (CDNA5 layouts, CDNA5 ISA §7.12.2)
// ---------------------------------------------------------------------------
// 16-bit A-matrix 16x32: lane m=lane&15; lanes<16 hold K 0..7 & 16..23,
// lanes>=16 hold K 8..15 & 24..31, two K values packed per VGPR.
__device__ __forceinline__ v16bf load_frag_bf16(const __bf16* tile, int stride, int lane) {
    const int m  = lane & 15;
    const int hi = (lane & 16) >> 1;        // 0 or 8
    const __bf16* p = tile + m * stride + hi;
    v16bf a;
#pragma unroll
    for (int j = 0; j < 8; ++j) {
        const int k = ((j >> 2) << 4) + ((j & 3) << 1);
        a[2 * j]     = p[k];
        a[2 * j + 1] = p[k + 1];
    }
    return a;
}

// 32-bit C/D 16x16: lane n=lane&15; VGPR r holds row r (lanes 0-15) / 8+r (lanes 16-31)
__device__ __forceinline__ void store_tile_f32(float* base, int stride, int lane, v8f d) {
    const int n  = lane & 15;
    const int mb = (lane & 16) >> 1;
#pragma unroll
    for (int r = 0; r < 8; ++r) base[(mb + r) * stride + n] = d[r];
}

// D = A(16xK) * B(KxN=16)^T-style: Y[m,n] = sum_k A[m,k] * W[n0+n, k]
__device__ __forceinline__ v8f gemm_tile(const __bf16* A, int sa,
                                         const __bf16* Wrows, int sw,
                                         int K, int lane) {
    v8f c = {};
#pragma unroll
    for (int k0 = 0; k0 < K; k0 += 32) {
        v16bf a = load_frag_bf16(A + k0, sa, lane);
        v16bf b = load_frag_bf16(Wrows + k0, sw, lane);
        c = __builtin_amdgcn_wmma_f32_16x16x32_bf16(false, a, false, b,
                                                    (short)0, c, false, false);
    }
    return c;
}

__device__ __forceinline__ float sigm(float x) { return 1.0f / (1.0f + __expf(-x)); }

// ---------------------------------------------------------------------------
// 1) fp32 -> bf16 weight conversion
// ---------------------------------------------------------------------------
__global__ void cvt_bf16_kernel(const float* __restrict__ src, __bf16* __restrict__ dst, int n) {
    int i = blockIdx.x * 256 + threadIdx.x;
    if (i < n) dst[i] = (__bf16)src[i];
}

// ---------------------------------------------------------------------------
// 2) top-3 stations per batch (the softmax support after the -1e9 mask)
// ---------------------------------------------------------------------------
__global__ void topk3_kernel(const float* __restrict__ l, int* __restrict__ sel) {
    int b = threadIdx.x;
    if (b >= BB) return;
    float v0 = -3.0e38f, v1 = -3.0e38f, v2 = -3.0e38f;
    int   i0 = 0, i1 = 0, i2 = 0;
    for (int n = 0; n < NN; ++n) {
        float v = l[b * NN + n];
        if (v > v0)      { v2 = v1; i2 = i1; v1 = v0; i1 = i0; v0 = v; i0 = n; }
        else if (v > v1) { v2 = v1; i2 = i1; v1 = v;  i1 = n; }
        else if (v > v2) { v2 = v;  i2 = n; }
    }
    sel[b * 3 + 0] = i0; sel[b * 3 + 1] = i1; sel[b * 3 + 2] = i2;
}

// ---------------------------------------------------------------------------
// 3) q_in[b,t,f] = sum_n l[b,n] * H[b,t,n,f]  -- the single full pass over H
// ---------------------------------------------------------------------------
__global__ __launch_bounds__(256) void qin_kernel(const float* __restrict__ H,
                                                  const float* __restrict__ l,
                                                  float* __restrict__ qin_f) {
    __shared__ float s_l[NN];
    const int bt = blockIdx.x;            // 0 .. B*T-1
    const int b  = bt / TT;
    const int f  = threadIdx.x;
    if (threadIdx.x < NN) s_l[threadIdx.x] = l[b * NN + threadIdx.x];
    __syncthreads();
    const float* __restrict__ Hp = H + (size_t)bt * NN * FF + f;
    float acc = 0.0f;
#pragma unroll 4
    for (int n = 0; n < NN; ++n) acc = fmaf(s_l[n], Hp[(size_t)n * FF], acc);
    qin_f[(size_t)bt * FF + f] = acc;
}

// ---------------------------------------------------------------------------
// 4) Fused masked attention for a tile of 16 timesteps (b fixed per block)
// ---------------------------------------------------------------------------
__global__ __launch_bounds__(256) void attn_kernel(
    const float* __restrict__ H, const float* __restrict__ qin_f,
    const int* __restrict__ sel,
    const __bf16* __restrict__ wq, const __bf16* __restrict__ wk,
    const __bf16* __restrict__ wv, const __bf16* __restrict__ wkk,
    const float* __restrict__ bq, const float* __restrict__ bk,
    const float* __restrict__ bv, const float* __restrict__ bkk,
    float* __restrict__ h_f, __bf16* __restrict__ h_bf) {

    __shared__ __bf16 s_qin[16 * FF];          // 8 KB  (A for q / first half of cat)
    __shared__ __bf16 s_hsel[3][16 * FF];      // 24 KB (A for keys/vals, 3 stations)
    __shared__ float  s_q[16 * FF];            // 16 KB
    __shared__ __bf16 s_keys[3][16 * FF];      // 24 KB
    __shared__ __bf16 s_hkn[16 * FF];          // 8 KB  (second half of cat)
    __shared__ float  s_scores[16 * 3];
    __shared__ float  s_attn[16 * 3];
    __shared__ int    s_sel[3];

    const int tile = blockIdx.x;               // 0..191
    const int b    = tile / 3;
    const int t0   = (tile % 3) * 16;
    const int row0 = b * TT + t0;              // global (b,t) row of first tile row
    const int tid  = threadIdx.x;
    const int lane = tid & 31;
    const int wv_id = tid >> 5;                // wave 0..7

    if (tid < 3) s_sel[tid] = sel[b * 3 + tid];
    __syncthreads();

    // --- stage LDS: q_in tile + 3 selected-station H tiles (fp32 -> bf16)
    for (int i = tid; i < 16 * FF; i += 256) {
        const int r = i >> 8, f = i & (FF - 1);
        s_qin[i] = (__bf16)qin_f[(size_t)(row0 + r) * FF + f];
#pragma unroll
        for (int k = 0; k < 3; ++k)
            s_hsel[k][i] = (__bf16)H[((size_t)(row0 + r) * NN + s_sel[k]) * FF + f];
    }
    __syncthreads();

    // --- q = q_in @ Wq^T + bq   (16 col-tiles over 8 waves)
#pragma unroll
    for (int i = 0; i < 2; ++i) {
        const int nb = wv_id * 2 + i;
        v8f c = gemm_tile(s_qin, FF, wq + nb * 16 * FF, FF, FF, lane);
        const int nG = nb * 16 + (lane & 15);
        const float bias = bq[nG];
#pragma unroll
        for (int r = 0; r < 8; ++r) c[r] += bias;
        store_tile_f32(s_q + nb * 16, FF, lane, c);
    }
    __syncthreads();

    // --- keys for 3 stations (48 tile jobs over 8 waves)
#pragma unroll
    for (int i = 0; i < 6; ++i) {
        const int job = wv_id * 6 + i;
        const int k   = job / 16;
        const int nb  = job % 16;
        v8f c = gemm_tile(s_hsel[k], FF, wk + nb * 16 * FF, FF, FF, lane);
        const int n  = lane & 15;
        const int mb = (lane & 16) >> 1;
        const float bias = bk[nb * 16 + n];
#pragma unroll
        for (int r = 0; r < 8; ++r)
            s_keys[k][(mb + r) * FF + nb * 16 + n] = (__bf16)(c[r] + bias);
    }
    __syncthreads();

    // --- scores + softmax over the 3 unmasked stations
    if (tid < 48) {
        const int t = tid / 3, k = tid % 3;
        float s = 0.0f;
        for (int f = 0; f < FF; ++f)
            s = fmaf(s_q[t * FF + f], (float)s_keys[k][t * FF + f], s);
        s_scores[tid] = s;
    }
    __syncthreads();
    if (tid < 16) {
        float a = s_scores[tid * 3], bs = s_scores[tid * 3 + 1], cs = s_scores[tid * 3 + 2];
        float m = fmaxf(a, fmaxf(bs, cs));
        float e0 = __expf(a - m), e1 = __expf(bs - m), e2 = __expf(cs - m);
        float inv = 1.0f / (e0 + e1 + e2);
        s_attn[tid * 3 + 0] = e0 * inv;
        s_attn[tid * 3 + 1] = e1 * inv;
        s_attn[tid * 3 + 2] = e2 * inv;
    }
    __syncthreads();

    // --- vals GEMM with attention weighting folded into D fragments
#pragma unroll
    for (int i = 0; i < 2; ++i) {
        const int nb = wv_id * 2 + i;
        const int n  = lane & 15;
        const int mb = (lane & 16) >> 1;
        const float bias_n = bv[nb * 16 + n];
        float acc[8];
#pragma unroll
        for (int r = 0; r < 8; ++r) acc[r] = 0.0f;
#pragma unroll
        for (int k = 0; k < 3; ++k) {
            v8f c = gemm_tile(s_hsel[k], FF, wv + nb * 16 * FF, FF, FF, lane);
#pragma unroll
            for (int r = 0; r < 8; ++r)
                acc[r] = fmaf(s_attn[(mb + r) * 3 + k], c[r] + bias_n, acc[r]);
        }
#pragma unroll
        for (int r = 0; r < 8; ++r)
            s_hkn[(mb + r) * FF + nb * 16 + n] = (__bf16)acc[r];
    }
    __syncthreads();

    // --- h = relu(cat(q_in, h_kn) @ Wkk^T + bkk), K = 512
#pragma unroll
    for (int i = 0; i < 2; ++i) {
        const int nb = wv_id * 2 + i;
        v8f c = {};
#pragma unroll
        for (int k0 = 0; k0 < 2 * FF; k0 += 32) {
            const __bf16* Ab = (k0 < FF) ? (s_qin + k0) : (s_hkn + (k0 - FF));
            v16bf a = load_frag_bf16(Ab, FF, lane);
            v16bf bfrag = load_frag_bf16(wkk + nb * 16 * (2 * FF) + k0, 2 * FF, lane);
            c = __builtin_amdgcn_wmma_f32_16x16x32_bf16(false, a, false, bfrag,
                                                        (short)0, c, false, false);
        }
        const int n  = lane & 15;
        const int mb = (lane & 16) >> 1;
        const int nG = nb * 16 + n;
        const float bias = bkk[nG];
#pragma unroll
        for (int r = 0; r < 8; ++r) {
            float vvv = c[r] + bias;
            vvv = vvv > 0.0f ? vvv : 0.0f;
            const size_t idx = (size_t)(row0 + mb + r) * FF + nG;
            h_f[idx]  = vvv;
            h_bf[idx] = (__bf16)vvv;
        }
    }
}

// ---------------------------------------------------------------------------
// 5) GI = X @ w_ih^T + b_ih for all (b,t) rows (per GRU layer)
// ---------------------------------------------------------------------------
__global__ __launch_bounds__(256) void gi_kernel(const __bf16* __restrict__ X,
                                                 const __bf16* __restrict__ w_ih,
                                                 const float* __restrict__ b_ih,
                                                 float* __restrict__ gi) {
    __shared__ __bf16 s_x[16 * FF];
    const int row0 = blockIdx.x * 16;
    const int tid  = threadIdx.x;
    const int lane = tid & 31;
    const int wv_id = tid >> 5;
    for (int i = tid; i < 16 * FF; i += 256) s_x[i] = X[(size_t)row0 * FF + i];
    __syncthreads();
#pragma unroll
    for (int i = 0; i < 6; ++i) {
        const int nb = wv_id * 6 + i;                   // 0..47 over N = 3F
        v8f c = gemm_tile(s_x, FF, w_ih + nb * 16 * FF, FF, FF, lane);
        const int n  = lane & 15;
        const int mb = (lane & 16) >> 1;
        const int nG = nb * 16 + n;
        const float bias = b_ih[nG];
#pragma unroll
        for (int r = 0; r < 8; ++r)
            gi[(size_t)(row0 + mb + r) * (3 * FF) + nG] = c[r] + bias;
    }
}

// ---------------------------------------------------------------------------
// 6) Sequential GRU scan (single block, WMMA gh per step)
// ---------------------------------------------------------------------------
__global__ __launch_bounds__(1024) void gru_scan_kernel(const float* __restrict__ gi,
                                                        const __bf16* __restrict__ w_hh,
                                                        const float* __restrict__ b_hh,
                                                        float* __restrict__ hstate,
                                                        float* __restrict__ ghbuf,
                                                        __bf16* __restrict__ y_bf,
                                                        int writeY) {
    __shared__ __bf16 s_h[BB * FF];                    // 32 KB, h in bf16 (A operand)
    const int tid  = threadIdx.x;
    const int lane = tid & 31;
    const int wv_id = tid >> 5;                        // 0..31

    for (int i = tid; i < BB * FF; i += 1024) { s_h[i] = (__bf16)0.0f; hstate[i] = 0.0f; }
    __syncthreads();

    for (int t = 0; t < TT; ++t) {
        // gh = h @ w_hh^T : 4 row-tiles x 48 col-tiles = 192 jobs over 32 waves
#pragma unroll
        for (int i = 0; i < 6; ++i) {
            const int job = wv_id * 6 + i;
            const int mt  = job / 48;
            const int nb  = job % 48;
            v8f c = gemm_tile(s_h + mt * 16 * FF, FF, w_hh + nb * 16 * FF, FF, FF, lane);
            store_tile_f32(ghbuf + (size_t)(mt * 16) * (3 * FF) + nb * 16, 3 * FF, lane, c);
        }
        __threadfence();
        __syncthreads();

        // GRU cell update: 64*256 elements over 1024 threads
        for (int i = tid; i < BB * FF; i += 1024) {
            const int b = i >> 8, f = i & (FF - 1);
            const size_t gir = (size_t)(b * TT + t) * (3 * FF);
            const size_t ghr = (size_t)b * (3 * FF);
            const float i_r = gi[gir + f];
            const float i_z = gi[gir + FF + f];
            const float i_n = gi[gir + 2 * FF + f];
            const float h_r = ghbuf[ghr + f] + b_hh[f];
            const float h_z = ghbuf[ghr + FF + f] + b_hh[FF + f];
            const float h_n = ghbuf[ghr + 2 * FF + f] + b_hh[2 * FF + f];
            const float h   = hstate[i];
            const float r   = sigm(i_r + h_r);
            const float z   = sigm(i_z + h_z);
            const float n   = tanhf(i_n + r * h_n);
            const float hn  = (1.0f - z) * n + z * h;
            hstate[i] = hn;
            s_h[i]    = (__bf16)hn;
            if (writeY) y_bf[(size_t)(b * TT + t) * FF + f] = (__bf16)hn;
        }
        __threadfence();
        __syncthreads();
    }
}

// ---------------------------------------------------------------------------
// 7) out = relu(h_last @ Wo^T + bo)  -> [B, 1]
// ---------------------------------------------------------------------------
__global__ void out_kernel(const float* __restrict__ hstate, const float* __restrict__ Wo,
                           const float* __restrict__ bo, float* __restrict__ out) {
    int b = threadIdx.x;
    if (b >= BB) return;
    float acc = 0.0f;
    for (int f = 0; f < FF; ++f) acc = fmaf(hstate[b * FF + f], Wo[f], acc);
    acc += bo[0];
    out[b] = acc > 0.0f ? acc : 0.0f;
}

// ---------------------------------------------------------------------------
// Host launch
// ---------------------------------------------------------------------------
extern "C" void kernel_launch(void* const* d_in, const int* in_sizes, int n_in,
                              void* d_out, int out_size, void* d_ws, size_t ws_size,
                              hipStream_t stream) {
    const float* H    = (const float*)d_in[0];
    const float* l    = (const float*)d_in[1];
    const float* Wk   = (const float*)d_in[2];
    const float* bk   = (const float*)d_in[3];
    const float* Wq   = (const float*)d_in[4];
    const float* bq   = (const float*)d_in[5];
    const float* Wv   = (const float*)d_in[6];
    const float* bv   = (const float*)d_in[7];
    const float* Wkk  = (const float*)d_in[8];
    const float* bkk  = (const float*)d_in[9];
    const float* gwih = (const float*)d_in[10];
    const float* gwhh = (const float*)d_in[11];
    const float* gbih = (const float*)d_in[12];
    const float* gbhh = (const float*)d_in[13];
    const float* Wo   = (const float*)d_in[14];
    const float* bo   = (const float*)d_in[15];
    float* out = (float*)d_out;

    // workspace layout
    char* ws = (char*)d_ws;
    size_t off = 0;
    auto alloc = [&](size_t bytes) -> void* {
        void* p = ws + off;
        off = (off + bytes + 255) & ~(size_t)255;
        return p;
    };
    int*    sel    = (int*)   alloc(BB * 3 * sizeof(int));
    __bf16* wq_bf  = (__bf16*)alloc((size_t)FF * FF * 2);
    __bf16* wk_bf  = (__bf16*)alloc((size_t)FF * FF * 2);
    __bf16* wv_bf  = (__bf16*)alloc((size_t)FF * FF * 2);
    __bf16* wkk_bf = (__bf16*)alloc((size_t)FF * 2 * FF * 2);
    __bf16* wih_bf = (__bf16*)alloc((size_t)LL * 3 * FF * FF * 2);
    __bf16* whh_bf = (__bf16*)alloc((size_t)LL * 3 * FF * FF * 2);
    float*  qin_f  = (float*) alloc((size_t)BB * TT * FF * 4);
    float*  h_f    = (float*) alloc((size_t)BB * TT * FF * 4);
    __bf16* h_bf   = (__bf16*)alloc((size_t)BB * TT * FF * 2);
    float*  gi_f   = (float*) alloc((size_t)BB * TT * 3 * FF * 4);
    __bf16* y_bf   = (__bf16*)alloc((size_t)BB * TT * FF * 2);
    float*  gh_f   = (float*) alloc((size_t)BB * 3 * FF * 4);
    float*  hstate = (float*) alloc((size_t)BB * FF * 4);
    (void)ws_size; (void)n_in; (void)in_sizes; (void)out_size;

    auto cvt = [&](const float* s, __bf16* d, int n) {
        cvt_bf16_kernel<<<(n + 255) / 256, 256, 0, stream>>>(s, d, n);
    };
    cvt(Wq,   wq_bf,  FF * FF);
    cvt(Wk,   wk_bf,  FF * FF);
    cvt(Wv,   wv_bf,  FF * FF);
    cvt(Wkk,  wkk_bf, FF * 2 * FF);
    cvt(gwih, wih_bf, LL * 3 * FF * FF);
    cvt(gwhh, whh_bf, LL * 3 * FF * FF);

    topk3_kernel<<<1, 64, 0, stream>>>(l, sel);

    qin_kernel<<<BB * TT, 256, 0, stream>>>(H, l, qin_f);

    attn_kernel<<<(BB * TT) / 16, 256, 0, stream>>>(H, qin_f, sel,
                                                    wq_bf, wk_bf, wv_bf, wkk_bf,
                                                    bq, bk, bv, bkk, h_f, h_bf);

    // GRU layer 0
    gi_kernel<<<(BB * TT) / 16, 256, 0, stream>>>(h_bf, wih_bf, gbih, gi_f);
    gru_scan_kernel<<<1, 1024, 0, stream>>>(gi_f, whh_bf, gbhh, hstate, gh_f, y_bf, 1);

    // GRU layer 1
    gi_kernel<<<(BB * TT) / 16, 256, 0, stream>>>(y_bf, wih_bf + (size_t)3 * FF * FF,
                                                  gbih + 3 * FF, gi_f);
    gru_scan_kernel<<<1, 1024, 0, stream>>>(gi_f, whh_bf + (size_t)3 * FF * FF,
                                            gbhh + 3 * FF, hstate, gh_f, y_bf, 0);

    out_kernel<<<1, 64, 0, stream>>>(hstate, Wo, bo, out);
}
